// KNNConv_35510789603930
// MI455X (gfx1250) — compile-verified, matrix operand order checked by reading, not run on hardware
//
#include <hip/hip_runtime.h>
#include <float.h>

// ---------------------------------------------------------------------------
// Problem constants (from reference): B=2, N=8192 -> BN=16384 points,
// K=20 neighbors, C_in=128, edge=2C=256, C_out=128.
// ---------------------------------------------------------------------------
#define NPTS    16384
#define KNN_K   20
#define C_IN    128
#define C_EDGE  256
#define C_OUT   128

typedef _Float16 v16h __attribute__((ext_vector_type(16)));
typedef _Float16 v8h  __attribute__((ext_vector_type(8)));
typedef float    v8f  __attribute__((ext_vector_type(8)));

// ---------------------------------------------------------------------------
// Kernel B tiling: each workgroup handles 8 points = 160 GEMM rows.
// LDS strides padded so stride/8 is odd -> ds_load_b128 A/B fetches hit 16
// distinct 16B bank groups per half-wave (conflict-free) and stay 16B aligned.
// ---------------------------------------------------------------------------
#define PTS_PER_BLK 8
#define ROWS        (PTS_PER_BLK * KNN_K)   // 160
#define MT_TILES    (ROWS / 16)             // 10
#define EDGE_STRIDE 264                     // 256 + 8 pad (f16): 528B rows
#define H1_STRIDE   136                     // 128 + 8 pad (f16): 272B rows

#define EDGE_OFF 0
#define H1_OFF   84480                      // 160*264*2
#define W1T_OFF  128000                     // + 160*136*2
#define W2T_OFF  195584                     // + 128*264*2
#define B1_OFF   230400                     // + 128*136*2
#define B2_OFF   230912
#define SMEM_BYTES 231424                   // ~226 KB of the 320 KB WGP LDS

// ---------------------------------------------------------------------------
// Kernel A: brute-force KNN (top-20 smallest squared distances, self incl.)
// One query per thread; candidates streamed through LDS; top-K kept entirely
// in registers (unrolled replace + rescan, so no dynamic register indexing).
// ---------------------------------------------------------------------------
__global__ void __launch_bounds__(256) knn_kernel(const float* __restrict__ pos,
                                                  int* __restrict__ knn) {
  __shared__ float spos[256 * 3];
  const int tid = threadIdx.x;
  const int qi  = blockIdx.x * 256 + tid;
  const float px = pos[qi * 3 + 0];
  const float py = pos[qi * 3 + 1];
  const float pz = pos[qi * 3 + 2];

  float bd[KNN_K];
  int   bi[KNN_K];
#pragma unroll
  for (int k = 0; k < KNN_K; ++k) { bd[k] = FLT_MAX; bi[k] = 0; }
  float cmax = FLT_MAX;   // current worst of the kept K
  int   cpos = 0;         // its slot

  for (int ch = 0; ch < NPTS / 256; ++ch) {
    const int j0 = ch * 256;
#pragma unroll
    for (int t = 0; t < 3; ++t)
      spos[tid + t * 256] = pos[j0 * 3 + tid + t * 256];
    __syncthreads();
#pragma unroll 4
    for (int jj = 0; jj < 256; ++jj) {
      const float dx = px - spos[jj * 3 + 0];
      const float dy = py - spos[jj * 3 + 1];
      const float dz = pz - spos[jj * 3 + 2];
      const float d2 = dx * dx + dy * dy + dz * dz;
      if (d2 < cmax) {
#pragma unroll
        for (int k = 0; k < KNN_K; ++k)
          if (k == cpos) { bd[k] = d2; bi[k] = j0 + jj; }
        cmax = bd[0]; cpos = 0;
#pragma unroll
        for (int k = 1; k < KNN_K; ++k)
          if (bd[k] > cmax) { cmax = bd[k]; cpos = k; }
      }
    }
    __syncthreads();
  }
#pragma unroll
  for (int k = 0; k < KNN_K; ++k) knn[qi * KNN_K + k] = bi[k];
}

// ---------------------------------------------------------------------------
// Kernel B helpers: 32B per-lane operand fetch as 2x 16B-aligned LDS loads.
// In this code's K order (same for A and B), lane l supplies 16 consecutive K values
// of row/col (l&15), starting at K = 16*(l>>4).
// ---------------------------------------------------------------------------
__device__ __forceinline__ v16h load_v16h(const _Float16* p) {
  v8h lo = *(const v8h*)p;
  v8h hi = *(const v8h*)(p + 8);
  return __builtin_shufflevector(lo, hi, 0, 1, 2, 3, 4, 5, 6, 7,
                                 8, 9, 10, 11, 12, 13, 14, 15);
}

// ---------------------------------------------------------------------------
// Kernel B: fused edge-feature build -> GEMM1(relu) -> GEMM2 -> max over K.
// Persistent grid; weights transposed into LDS once per WG, B operands held
// in registers across the whole block loop. h2 (f32) aliases the dead edge
// buffer between barriers.
// ---------------------------------------------------------------------------
__global__ void __launch_bounds__(256) edgeconv_kernel(
    const float* __restrict__ feat, const int* __restrict__ knn,
    const float* __restrict__ W1, const float* __restrict__ b1,
    const float* __restrict__ W2, const float* __restrict__ b2,
    float* __restrict__ out) {
  extern __shared__ char smem[];
  _Float16* edge = (_Float16*)(smem + EDGE_OFF);
  float*    h2   = (float*)(smem + EDGE_OFF);   // aliases edge (160*128*4 <= 84480)
  _Float16* h1   = (_Float16*)(smem + H1_OFF);
  _Float16* w1t  = (_Float16*)(smem + W1T_OFF); // W1^T: [out=128][in=256], stride 264
  _Float16* w2t  = (_Float16*)(smem + W2T_OFF); // W2^T: [out=128][in=128], stride 136
  float* sb1 = (float*)(smem + B1_OFF);
  float* sb2 = (float*)(smem + B2_OFF);

  const int tid  = threadIdx.x;
  const int lane = tid & 31;
  const int wave = tid >> 5;

  // one-time: transpose+downconvert weights, stage biases
  for (int i = tid; i < C_EDGE * C_OUT; i += 256) {
    const int kk = i >> 7, n = i & 127;             // W1[kk][n]
    w1t[n * EDGE_STRIDE + kk] = (_Float16)W1[i];
  }
  for (int i = tid; i < C_OUT * C_OUT; i += 256) {
    const int kk = i >> 7, n = i & 127;             // W2[kk][n]
    w2t[n * H1_STRIDE + kk] = (_Float16)W2[i];
  }
  if (tid < 128) sb1[tid] = b1[tid];
  else           sb2[tid - 128] = b2[tid - 128];
  __syncthreads();

  // per-wave constants: each wave owns one 16-wide output-channel tile
  const int nt    = wave;
  const int ncol  = nt * 16 + (lane & 15);   // this lane's output channel
  const int khal  = (lane >> 4) * 16;        // K start of this lane's 16-K run (A and B alike)
  const int rhalf = (lane >> 4) * 8;         // C/D row offset per layout
  v16h B1r[8];
#pragma unroll
  for (int ks = 0; ks < 8; ++ks)
    B1r[ks] = load_v16h(w1t + ncol * EDGE_STRIDE + ks * 32 + khal);
  v16h B2r[4];
#pragma unroll
  for (int ks = 0; ks < 4; ++ks)
    B2r[ks] = load_v16h(w2t + ncol * H1_STRIDE + ks * 32 + khal);
  const float b1v = sb1[ncol];
  const float b2v = sb2[ncol];

  const int nblk = NPTS / PTS_PER_BLK;   // 2048
  for (int blk = blockIdx.x; blk < nblk; blk += gridDim.x) {
    const int g0 = blk * PTS_PER_BLK;

    // ---- build edge features [160 x 256] f16 in LDS ----
    {
      const int c  = tid;          // edge-feature column, fixed per thread
      const int cl = c & 127;
      int pt = 0, kk2 = 0;
      float fcen = feat[g0 * C_IN + cl];
      for (int r = 0; r < ROWS; ++r) {
        float v;
        if (c < C_IN) {
          v = fcen;                                   // center part
        } else {
          const int nb = knn[(g0 + pt) * KNN_K + kk2]; // uniform per wave
          v = feat[nb * C_IN + cl] - fcen;             // neighbor - center
        }
        edge[r * EDGE_STRIDE + c] = (_Float16)v;
        if (++kk2 == KNN_K) {
          kk2 = 0; ++pt;
          if (pt < PTS_PER_BLK) fcen = feat[(g0 + pt) * C_IN + cl];
        }
      }
    }
    __syncthreads();

    // ---- GEMM1: h1 = relu(edge @ W1 + b1), [160x256]x[256x128] ----
    for (int mt = 0; mt < MT_TILES; ++mt) {
      v8f acc = {};
      const _Float16* arow =
          edge + (mt * 16 + (lane & 15)) * EDGE_STRIDE + khal;
#pragma unroll
      for (int ks = 0; ks < 8; ++ks) {
        v16h A = load_v16h(arow + ks * 32);
        acc = __builtin_amdgcn_wmma_f32_16x16x32_f16(false, A, false, B1r[ks],
                                                     (short)0, acc, false, false);
      }
#pragma unroll
      for (int v = 0; v < 8; ++v) {
        float x = acc[v] + b1v;
        x = fmaxf(x, 0.0f);
        h1[(mt * 16 + rhalf + v) * H1_STRIDE + ncol] = (_Float16)x;
      }
    }
    __syncthreads();   // h1 ready; edge dead -> safe to overwrite with h2

    // ---- GEMM2: h2 = h1 @ W2 + b2, [160x128]x[128x128] ----
    for (int mt = 0; mt < MT_TILES; ++mt) {
      v8f acc = {};
      const _Float16* arow = h1 + (mt * 16 + (lane & 15)) * H1_STRIDE + khal;
#pragma unroll
      for (int ks = 0; ks < 4; ++ks) {
        v16h A = load_v16h(arow + ks * 32);
        acc = __builtin_amdgcn_wmma_f32_16x16x32_f16(false, A, false, B2r[ks],
                                                     (short)0, acc, false, false);
      }
#pragma unroll
      for (int v = 0; v < 8; ++v)
        h2[(mt * 16 + rhalf + v) * C_OUT + ncol] = acc[v] + b2v;
    }
    __syncthreads();

    // ---- max over K neighbors -> global output ----
    for (int e = tid; e < PTS_PER_BLK * C_OUT; e += 256) {
      const int pt = e >> 7, chn = e & 127;
      float m = -FLT_MAX;
#pragma unroll
      for (int k = 0; k < KNN_K; ++k)
        m = fmaxf(m, h2[(pt * KNN_K + k) * C_OUT + chn]);
      out[(g0 + pt) * C_OUT + chn] = m;
    }
    __syncthreads();   // h2 read done before next block's edge build
  }
}

// ---------------------------------------------------------------------------
// Launch: KNN -> fused EdgeConv MLP. knn indices live in d_ws (16384*20 ints).
// ---------------------------------------------------------------------------
extern "C" void kernel_launch(void* const* d_in, const int* in_sizes, int n_in,
                              void* d_out, int out_size, void* d_ws, size_t ws_size,
                              hipStream_t stream) {
  (void)in_sizes; (void)n_in; (void)out_size; (void)ws_size;
  const float* pos  = (const float*)d_in[0];
  const float* feat = (const float*)d_in[1];
  const float* W1   = (const float*)d_in[2];
  const float* b1   = (const float*)d_in[3];
  const float* W2   = (const float*)d_in[4];
  const float* b2   = (const float*)d_in[5];
  float* out = (float*)d_out;
  int*   knn = (int*)d_ws;

  knn_kernel<<<NPTS / 256, 256, 0, stream>>>(pos, knn);

  (void)hipFuncSetAttribute(reinterpret_cast<const void*>(&edgeconv_kernel),
                            hipFuncAttributeMaxDynamicSharedMemorySize,
                            SMEM_BYTES);
  edgeconv_kernel<<<512, 256, SMEM_BYTES, stream>>>(feat, knn, W1, b1, W2, b2,
                                                    out);
}
